// HausdorffLoss_65524021067918
// MI455X (gfx1250) — matrix operand order repaired and checked by reading, not compiled, not run on hardware
//
#include <hip/hip_runtime.h>
#include <stdint.h>

typedef __attribute__((ext_vector_type(16))) __bf16 v16bf;
typedef __attribute__((ext_vector_type(8)))  __bf16 v8bf;
typedef __attribute__((ext_vector_type(8)))  float  v8f;

#define N_PTS 4096
#define M_PTS 4096
#define DIM   1024
#define KBLK  32
#define LDS_STRIDE 40   // 32 bf16 payload + 8 pad = 80 B/row (16B aligned, conflict-free: 20*l mod 64)

#if defined(__has_builtin)
#  if __has_builtin(__builtin_amdgcn_global_load_async_to_lds_b128)
#    define HAS_ASYNC_LDS 1
#  endif
#endif
#ifndef HAS_ASYNC_LDS
#  define HAS_ASYNC_LDS 0
#endif

#if !HAS_ASYNC_LDS && defined(__AMDGCN__)
#warning "CDNA5 async-to-LDS builtin unavailable: using sync load+ds_store fallback"
#endif

// Pointer types matching the builtin's signature (from hipcc diagnostic:
// param is 'int __attribute__((vector_size(16))) __device__ *', i.e. as(1) v4i*)
typedef int async_v4i __attribute__((vector_size(16)));
typedef __attribute__((address_space(1))) async_v4i* async_gptr;
typedef __attribute__((address_space(3))) async_v4i* async_lptr;

union Frag {
    v16bf v;
    v8bf  h[2];
};

__device__ __forceinline__ uint32_t f32_to_bf16_rne(float f) {
    uint32_t b = __float_as_uint(f);
    return (b + 0x7FFFu + ((b >> 16) & 1u)) >> 16;
}

// 16-byte global -> LDS copy (async on CDNA5; ASYNCcnt-tracked, no VGPR data)
__device__ __forceinline__ void g2lds_b128(const unsigned short* g, unsigned short* l) {
#if HAS_ASYNC_LDS
    __builtin_amdgcn_global_load_async_to_lds_b128(
        (async_gptr)(uintptr_t)g,
        (async_lptr)(uint32_t)(uintptr_t)l,
        0, 0);
#else
    *(uint4*)l = *(const uint4*)g;
#endif
}

__device__ __forceinline__ void wait_async_all() {
#if HAS_ASYNC_LDS
    asm volatile("s_wait_asynccnt 0" ::: "memory");
#endif
}

// ---------------------------------------------------------------------------
// Kernel 1: per-row norms + f32 -> (hi,lo) bf16 split (done ONCE) + min init
// ---------------------------------------------------------------------------
__global__ __launch_bounds__(256)
void hd_prep(const float* __restrict__ set1, const float* __restrict__ set2,
             unsigned short* __restrict__ xhi, unsigned short* __restrict__ xlo,
             unsigned short* __restrict__ yhi, unsigned short* __restrict__ ylo,
             float* __restrict__ x2, float* __restrict__ y2,
             int* __restrict__ rowmin, int* __restrict__ colmin) {
    __shared__ float red[256];
    const int r = blockIdx.x;
    const bool isA = (r < N_PTS);
    const float* src = isA ? (set1 + (size_t)r * DIM) : (set2 + (size_t)(r - N_PTS) * DIM);
    unsigned short* dhi = isA ? (xhi + (size_t)r * DIM) : (yhi + (size_t)(r - N_PTS) * DIM);
    unsigned short* dlo = isA ? (xlo + (size_t)r * DIM) : (ylo + (size_t)(r - N_PTS) * DIM);

    const int t = threadIdx.x;
    const float4 v = *(const float4*)(src + t * 4);
    uint32_t h0 = f32_to_bf16_rne(v.x), h1 = f32_to_bf16_rne(v.y);
    uint32_t h2 = f32_to_bf16_rne(v.z), h3 = f32_to_bf16_rne(v.w);
    uint32_t l0 = f32_to_bf16_rne(v.x - __uint_as_float(h0 << 16));
    uint32_t l1 = f32_to_bf16_rne(v.y - __uint_as_float(h1 << 16));
    uint32_t l2 = f32_to_bf16_rne(v.z - __uint_as_float(h2 << 16));
    uint32_t l3 = f32_to_bf16_rne(v.w - __uint_as_float(h3 << 16));
    *(uint2*)(dhi + t * 4) = make_uint2(h0 | (h1 << 16), h2 | (h3 << 16));
    *(uint2*)(dlo + t * 4) = make_uint2(l0 | (l1 << 16), l2 | (l3 << 16));

    float s = v.x * v.x + v.y * v.y + v.z * v.z + v.w * v.w;
    red[t] = s;
    __syncthreads();
    for (int off = 128; off > 0; off >>= 1) {
        if (t < off) red[t] += red[t + off];
        __syncthreads();
    }
    if (t == 0) {
        if (isA) { x2[r] = red[0]; rowmin[r] = 0x7F800000; }
        else     { y2[r - N_PTS] = red[0]; colmin[r - N_PTS] = 0x7F800000; }
    }
}

// ---------------------------------------------------------------------------
// Kernel 2: split-bf16 WMMA GEMM + fused distance + row/col mins
// Block 256 thr (8 waves), 128x128 output tile; wave = 32x64 (2x4 WMMA accs).
// K staged in 32-wide slices, double-buffered via async-to-LDS.
// ---------------------------------------------------------------------------
__global__ __launch_bounds__(256)
void hd_gemm(const unsigned short* __restrict__ xhi, const unsigned short* __restrict__ xlo,
             const unsigned short* __restrict__ yhi, const unsigned short* __restrict__ ylo,
             const float* __restrict__ x2, const float* __restrict__ y2,
             int* __restrict__ rowmin, int* __restrict__ colmin) {
    __shared__ __align__(16) unsigned short lAhi[2][128 * LDS_STRIDE];
    __shared__ __align__(16) unsigned short lAlo[2][128 * LDS_STRIDE];
    __shared__ __align__(16) unsigned short lBhi[2][128 * LDS_STRIDE];
    __shared__ __align__(16) unsigned short lBlo[2][128 * LDS_STRIDE];

    const int nBase = blockIdx.y * 128;
    const int mBase = blockIdx.x * 128;

    const int t    = threadIdx.x;
    const int lane = t & 31;
    const int w    = t >> 5;
    const int wr   = w >> 1;     // 0..3 (row group of 32)
    const int wc   = w & 1;      // 0..1 (col group of 64)
    const int half = lane >> 4;
    const int l15  = lane & 15;

    // Per-thread async-copy slots: 8 x 16B per k-step (2 each of Ahi/Alo/Bhi/Blo)
    const unsigned short* gbase[8];
    int ldsOff[8];
#pragma unroll
    for (int it = 0; it < 8; ++it) {
        const int cc  = ((it & 1) << 8) + t;   // 0..511
        const int row = cc >> 2;               // 0..127
        const int c4  = cc & 3;                // 16B chunk in 64B row
        ldsOff[it] = row * LDS_STRIDE + c4 * 8;
        const size_t ga = (size_t)(nBase + row) * DIM + c4 * 8;
        const size_t gb = (size_t)(mBase + row) * DIM + c4 * 8;
        gbase[it] = (it < 2) ? (xhi + ga) : (it < 4) ? (xlo + ga)
                  : (it < 6) ? (yhi + gb) : (ylo + gb);
    }

    auto issue_stage = [&](int kb, int buf) {
#pragma unroll
        for (int it = 0; it < 8; ++it) {
            unsigned short* lb = (it < 2) ? &lAhi[buf][0] : (it < 4) ? &lAlo[buf][0]
                               : (it < 6) ? &lBhi[buf][0] : &lBlo[buf][0];
            g2lds_b128(gbase[it] + kb, lb + ldsOff[it]);
        }
    };

    v8f acc[2][4];
#pragma unroll
    for (int i = 0; i < 2; ++i)
#pragma unroll
        for (int j = 0; j < 4; ++j)
            acc[i][j] = (v8f){0.f, 0.f, 0.f, 0.f, 0.f, 0.f, 0.f, 0.f};

    issue_stage(0, 0);

    const int NSTEP = DIM / KBLK;   // 32
    for (int i = 0; i < NSTEP; ++i) {
        const int cur = i & 1;
        wait_async_all();
        __syncthreads();                       // buf[cur] ready for all waves
        if (i + 1 < NSTEP) issue_stage((i + 1) * KBLK, 1 - cur);

        // ---- fragments from LDS ----
        // A (16-bit 16x32): lane holds M=l15, K chunks {k0..k0+7, k0+16..k0+23}, k0=half*8
        Frag aHi[2], aLo[2], bHi[4], bLo[4];
#pragma unroll
        for (int ti = 0; ti < 2; ++ti) {
            const int base = (wr * 32 + ti * 16 + l15) * LDS_STRIDE + half * 8;
            aHi[ti].h[0] = *(const v8bf*)&lAhi[cur][base];
            aHi[ti].h[1] = *(const v8bf*)&lAhi[cur][base + 16];
            aLo[ti].h[0] = *(const v8bf*)&lAlo[cur][base];
            aLo[ti].h[1] = *(const v8bf*)&lAlo[cur][base + 16];
        }
        // B (16-bit 32x16): lane holds N=l15, K chunks {k0..k0+7, k0+16..k0+23}, k0=half*8
#pragma unroll
        for (int tj = 0; tj < 4; ++tj) {
            const int base = (wc * 64 + tj * 16 + l15) * LDS_STRIDE + half * 8;
            bHi[tj].h[0] = *(const v8bf*)&lBhi[cur][base];
            bHi[tj].h[1] = *(const v8bf*)&lBhi[cur][base + 16];
            bLo[tj].h[0] = *(const v8bf*)&lBlo[cur][base];
            bLo[tj].h[1] = *(const v8bf*)&lBlo[cur][base + 16];
        }

        // ---- 3-term split: xy ~= lo*hi + hi*lo + hi*hi ----
#pragma unroll
        for (int ti = 0; ti < 2; ++ti)
#pragma unroll
            for (int tj = 0; tj < 4; ++tj) {
                v8f c = acc[ti][tj];
                c = __builtin_amdgcn_wmma_f32_16x16x32_bf16(
                        false, aLo[ti].v, false, bHi[tj].v, (short)0, c, false, false);
                c = __builtin_amdgcn_wmma_f32_16x16x32_bf16(
                        false, aHi[ti].v, false, bLo[tj].v, (short)0, c, false, false);
                c = __builtin_amdgcn_wmma_f32_16x16x32_bf16(
                        false, aHi[ti].v, false, bHi[tj].v, (short)0, c, false, false);
                acc[ti][tj] = c;
            }
        // single barrier/iter: buf[cur] is only overwritten at step i+1,
        // after every wave passed the barrier at the top of step i+1.
    }

    // ---- epilogue: dist = (x2 + y2 - 2*xy)/D, fold into mins ----
    const float invD = 1.0f / (float)DIM;
#pragma unroll
    for (int ti = 0; ti < 2; ++ti) {
        const int nTile = nBase + wr * 32 + ti * 16;
#pragma unroll
        for (int tj = 0; tj < 4; ++tj) {
            const int mTile = mBase + wc * 64 + tj * 16;
            const int m     = mTile + l15;
            const float y2m = y2[m];
            float cmin = __int_as_float(0x7F800000);
#pragma unroll
            for (int v = 0; v < 8; ++v) {
                const int n = nTile + v + half * 8;
                float d = (x2[n] + y2m - 2.0f * acc[ti][tj][v]) * invD;
                d = fmaxf(d, 0.0f);          // non-negative -> int order == float order
                cmin = fminf(cmin, d);
                float r = d;                 // reduce across 16 lanes sharing n
                r = fminf(r, __shfl_xor(r, 1));
                r = fminf(r, __shfl_xor(r, 2));
                r = fminf(r, __shfl_xor(r, 4));
                r = fminf(r, __shfl_xor(r, 8));
                if (l15 == 0) atomicMin(&rowmin[n], __float_as_int(r));
            }
            atomicMin(&colmin[m], __float_as_int(cmin));
        }
    }
}

// ---------------------------------------------------------------------------
// Kernel 3: final reduction -> scalar loss
// ---------------------------------------------------------------------------
__global__ __launch_bounds__(256)
void hd_finish(const int* __restrict__ rowmin, const int* __restrict__ colmin,
               float* __restrict__ out) {
    __shared__ float red[256];
    float s = 0.0f;
    for (int i = threadIdx.x; i < N_PTS; i += 256) s += __int_as_float(rowmin[i]);
    for (int i = threadIdx.x; i < M_PTS; i += 256) s += __int_as_float(colmin[i]);
    red[threadIdx.x] = s;
    __syncthreads();
    for (int off = 128; off > 0; off >>= 1) {
        if ((int)threadIdx.x < off) red[threadIdx.x] += red[threadIdx.x + off];
        __syncthreads();
    }
    if (threadIdx.x == 0)
        out[0] = red[0] * (0.5f / 4096.0f);   // (sum_row/N + sum_col/M) * 0.5
}

// ---------------------------------------------------------------------------
extern "C" void kernel_launch(void* const* d_in, const int* in_sizes, int n_in,
                              void* d_out, int out_size, void* d_ws, size_t ws_size,
                              hipStream_t stream) {
    const float* set1 = (const float*)d_in[0];
    const float* set2 = (const float*)d_in[1];

    unsigned short* xhi = (unsigned short*)d_ws;               // [4096*1024] each
    unsigned short* xlo = xhi + (size_t)N_PTS * DIM;
    unsigned short* yhi = xlo + (size_t)N_PTS * DIM;
    unsigned short* ylo = yhi + (size_t)M_PTS * DIM;
    float* x2     = (float*)(ylo + (size_t)M_PTS * DIM);
    float* y2     = x2 + N_PTS;
    int*   rowmin = (int*)(y2 + M_PTS);
    int*   colmin = rowmin + N_PTS;

    hd_prep<<<N_PTS + M_PTS, 256, 0, stream>>>(set1, set2, xhi, xlo, yhi, ylo,
                                               x2, y2, rowmin, colmin);

    dim3 grid(M_PTS / 128, N_PTS / 128);   // 32 x 32 blocks
    hd_gemm<<<grid, 256, 0, stream>>>(xhi, xlo, yhi, ylo, x2, y2, rowmin, colmin);

    hd_finish<<<1, 256, 0, stream>>>(rowmin, colmin, (float*)d_out);
}
